// CasualSelfAttention_34359739049
// MI455X (gfx1250) — matrix operand order repaired and checked by reading, not compiled, not run on hardware
//
#include <hip/hip_runtime.h>
#include <math.h>

#define B_ 4
#define T_ 2048
#define C_ 1024
#define H_ 16
#define D_ 64
#define M_ (B_*T_)   // 8192 rows

typedef __attribute__((ext_vector_type(16))) __bf16 v16bf;
typedef __attribute__((ext_vector_type(8)))  float  v8f;

union Frag { v16bf v; unsigned int u[8]; };

__device__ __forceinline__ unsigned short f2bf(float f) {
  unsigned int x = __float_as_uint(f);
  x += 0x7fffu + ((x >> 16) & 1u);          // round-to-nearest-even
  return (unsigned short)(x >> 16);
}

// A-fragment (16x32 bf16): lane(0-15)=row, VGPR p holds K pair starting at:
__device__ __forceinline__ int a_kb(int p, int hi) { return 2*p + ((p >= 4) ? 8 : 0) + (hi ? 8 : 0); }
// B-fragment (32x16 bf16): lane%16=column N, VGPR p holds K pair starting at:
__device__ __forceinline__ int b_kb(int p, int hi) { return 2*p + ((p >= 4) ? 8 : 0) + (hi ? 8 : 0); }

// Low 32 bits of a generic pointer to LDS == byte offset within the LDS
// allocation (flat LDS aperture keeps the offset in addr[31:0]).
__device__ __forceinline__ unsigned lds_off(const void* p) {
  return (unsigned)(unsigned long long)p;
}

// CDNA5 async DMA: global -> LDS, 16 bytes per lane, tracked by ASYNCcnt.
__device__ __forceinline__ void async_copy_b128(unsigned lds_byte, const void* gbase,
                                                unsigned gbyte_off) {
  asm volatile("global_load_async_to_lds_b128 %0, %1, %2"
               :: "v"(lds_byte), "v"(gbyte_off),
                  "s"((unsigned long long)(size_t)gbase)
               : "memory");
}
__device__ __forceinline__ void wait_async0() {
  asm volatile("s_wait_asynccnt 0" ::: "memory");
}

// ---------------------------------------------------------------------------
// Pre-pass 1: f32 -> bf16 elementwise (x)
// ---------------------------------------------------------------------------
__global__ __launch_bounds__(256)
void convert_kernel(const float* __restrict__ in, unsigned short* __restrict__ out, int n) {
  int i = blockIdx.x * 256 + threadIdx.x;
  if (i < n) out[i] = f2bf(in[i]);
}

// ---------------------------------------------------------------------------
// Pre-pass 2: W [K][N] f32 -> Wt [N][K] bf16 (tiled transpose via LDS)
// ---------------------------------------------------------------------------
__global__ __launch_bounds__(256)
void transpose_w_kernel(const float* __restrict__ W, unsigned short* __restrict__ Wt) {
  __shared__ unsigned short Ts[64][65];
  const int k0 = blockIdx.y * 64, n0 = blockIdx.x * 64;
  for (int i = threadIdx.x; i < 64 * 64; i += 256) {
    int r = i >> 6, c = i & 63;                 // r: k offset, c: n offset
    Ts[c][r] = f2bf(W[(size_t)(k0 + r) * C_ + n0 + c]);
  }
  __syncthreads();
  for (int i = threadIdx.x; i < 64 * 64; i += 256) {
    int r = i >> 6, c = i & 63;                 // r: n offset, c: k offset
    Wt[(size_t)(n0 + r) * C_ + k0 + c] = Ts[r][c];
  }
}

// ---------------------------------------------------------------------------
// bf16 GEMM: out = A[M_,C_] * Bt[C_,C_]^T + bias  (Bt stored [n][k])
// 256 threads (8 waves), block tile 128x128, K-step 32, double-buffered
// async global->LDS staging. Wave (w&3, w>>2) computes a 32x64 sub-tile.
// out_mode 0: bf16 [B,H,T,D]; 1: bf16 [B,H,D,T]; 2: f32 [M_,C_].
// ---------------------------------------------------------------------------
__device__ __forceinline__ void stage_ab(int tid, unsigned asB, unsigned bsB,
                                         const unsigned short* A, const unsigned short* Bt,
                                         int mBase, int nBase, int k0) {
  // 512 B128 chunks each for A and B tiles (128 rows x 4 segs), 2+2 per thread
  int c = tid, r = c >> 2, s = c & 3;
  async_copy_b128(asB + r * 80 + s * 16, A,  (unsigned)(((mBase + r) * C_ + k0 + s * 8) * 2));
  async_copy_b128(bsB + r * 80 + s * 16, Bt, (unsigned)(((nBase + r) * C_ + k0 + s * 8) * 2));
  c = tid + 256; r = c >> 2; s = c & 3;
  async_copy_b128(asB + r * 80 + s * 16, A,  (unsigned)(((mBase + r) * C_ + k0 + s * 8) * 2));
  async_copy_b128(bsB + r * 80 + s * 16, Bt, (unsigned)(((nBase + r) * C_ + k0 + s * 8) * 2));
}

__global__ __launch_bounds__(256)
void gemm_bf16_kernel(const unsigned short* __restrict__ A,
                      const unsigned short* __restrict__ Bt,
                      const float* __restrict__ bias,
                      unsigned short* __restrict__ out_bf,
                      float* __restrict__ out_f, int out_mode)
{
  __shared__ __align__(16) unsigned short As[2][128][40];   // 80B row stride (16B-aligned)
  __shared__ __align__(16) unsigned short Bs[2][128][40];

  const int tid  = threadIdx.x;
  const int lane = tid & 31;
  const int wave = tid >> 5;
  const int lh   = lane & 15;
  const int hi   = lane >> 4;
  const int mBase = blockIdx.y * 128;
  const int nBase = blockIdx.x * 128;
  const int m0 = (wave & 3) * 32;       // wave's rows within block tile
  const int n0 = (wave >> 2) * 64;      // wave's cols within block tile

  const unsigned asBase = lds_off(&As[0][0][0]);
  const unsigned bsBase = lds_off(&Bs[0][0][0]);

  v8f acc[2][4];
  for (int mt = 0; mt < 2; ++mt)
    for (int nt = 0; nt < 4; ++nt)
      for (int r = 0; r < 8; ++r) acc[mt][nt][r] = 0.0f;

  stage_ab(tid, asBase, bsBase, A, Bt, mBase, nBase, 0);   // prologue -> buf 0

  for (int kk = 0; kk < C_ / 32; ++kk) {
    const int buf = kk & 1;
    wait_async0();        // our chunks of buf `buf` are resident
    __syncthreads();      // everyone's chunks are resident; prev reads done
    if (kk + 1 < C_ / 32)
      stage_ab(tid, asBase + (buf ^ 1) * 10240, bsBase + (buf ^ 1) * 10240,
               A, Bt, mBase, nBase, (kk + 1) * 32);

    Frag a[2];
#pragma unroll
    for (int mt = 0; mt < 2; ++mt)
#pragma unroll
      for (int p = 0; p < 8; ++p)
        a[mt].u[p] = *(const unsigned int*)&As[buf][m0 + mt * 16 + lh][a_kb(p, hi)];

#pragma unroll
    for (int nt = 0; nt < 4; ++nt) {
      Frag b;
#pragma unroll
      for (int p = 0; p < 8; ++p)
        b.u[p] = *(const unsigned int*)&Bs[buf][n0 + nt * 16 + lh][b_kb(p, hi)];
#pragma unroll
      for (int mt = 0; mt < 2; ++mt)
        acc[mt][nt] = __builtin_amdgcn_wmma_f32_16x16x32_bf16(
            false, a[mt].v, false, b.v, (short)0, acc[mt][nt], false, false);
    }
  }

  // Epilogue: bias + scatter. C/D layout: row = reg + 8*hi, col = lane%16.
#pragma unroll
  for (int mt = 0; mt < 2; ++mt) {
#pragma unroll
    for (int nt = 0; nt < 4; ++nt) {
      const int n  = nBase + n0 + nt * 16 + lh;
      const float bv = bias[n];
#pragma unroll
      for (int r = 0; r < 8; ++r) {
        const int m  = mBase + m0 + mt * 16 + r + hi * 8;
        const float val = acc[mt][nt][r] + bv;
        if (out_mode == 2) {
          out_f[(size_t)m * C_ + n] = val;
        } else {
          const int bb = m >> 11;        // m / T_
          const int t  = m & (T_ - 1);
          const int h  = n >> 6;         // n / D_
          const int d  = n & (D_ - 1);
          size_t idx;
          if (out_mode == 0)
            idx = ((size_t)(bb * H_ + h) * T_ + t) * D_ + d;   // [B,H,T,D]
          else
            idx = ((size_t)(bb * H_ + h) * D_ + d) * T_ + t;   // [B,H,D,T]
          out_bf[idx] = f2bf(val);
        }
      }
    }
  }
}

// ---------------------------------------------------------------------------
// Flash attention: block = 4 waves over a 64-row q-block of one (b,h).
// K/V tiles (32 keys) staged into shared LDS via async B128 copies,
// double-buffered; all matmuls via v_wmma_f32_16x16x32_bf16.
// ---------------------------------------------------------------------------
__device__ __forceinline__ void stage_kv(int tid, unsigned ksB, unsigned vsB,
                                         const unsigned short* kbp, const unsigned short* vbp,
                                         int kb0) {
  // K tile: 32 rows x 64 d  = 256 B128 chunks (8/row)
  int c = tid, r = c >> 3, s = c & 7;
  async_copy_b128(ksB + r * 144 + s * 16, kbp, (unsigned)(((kb0 + r) * D_ + s * 8) * 2));
  c = tid + 128; r = c >> 3; s = c & 7;
  async_copy_b128(ksB + r * 144 + s * 16, kbp, (unsigned)(((kb0 + r) * D_ + s * 8) * 2));
  // V^T tile: 64 d-rows x 32 keys = 256 B128 chunks (4/row)
  c = tid; r = c >> 2; s = c & 3;
  async_copy_b128(vsB + r * 80 + s * 16, vbp, (unsigned)((r * T_ + kb0 + s * 8) * 2));
  c = tid + 128; r = c >> 2; s = c & 3;
  async_copy_b128(vsB + r * 80 + s * 16, vbp, (unsigned)((r * T_ + kb0 + s * 8) * 2));
}

__global__ __launch_bounds__(128)
void attn_kernel(const unsigned short* __restrict__ q,
                 const unsigned short* __restrict__ k,
                 const unsigned short* __restrict__ vt,
                 unsigned short* __restrict__ y)
{
  __shared__ __align__(16) unsigned short Ks[2][32][72];    // [key][d], 144B rows
  __shared__ __align__(16) unsigned short Vs[2][64][40];    // [d][key], 80B rows
  __shared__ __align__(16) unsigned short pLds[4][16][36];  // per-wave P transpose

  const int tid  = threadIdx.x;
  const int lane = tid & 31;
  const int wave = tid >> 5;
  const int lh   = lane & 15;
  const int hi   = lane >> 4;

  const int blk    = blockIdx.x;
  const int bh     = blk >> 5;
  const int tBlock = (blk & 31) << 6;
  const int tTile  = tBlock + wave * 16;

  const unsigned short* qb = q  + (size_t)bh * T_ * D_;
  const unsigned short* kb = k  + (size_t)bh * T_ * D_;
  const unsigned short* vb = vt + (size_t)bh * D_ * T_;

  const unsigned ksBase = lds_off(&Ks[0][0][0]);
  const unsigned vsBase = lds_off(&Vs[0][0][0]);

  // Q fragments: 16 x 64 as two A-fragments (K chunks of 32)
  Frag qf[2];
#pragma unroll
  for (int c = 0; c < 2; ++c)
#pragma unroll
    for (int p = 0; p < 8; ++p)
      qf[c].u[p] = *(const unsigned int*)(qb + (size_t)(tTile + lh) * D_ + c * 32 + a_kb(p, hi));

  float mrow[8], lrow[8];
  v8f oacc[4];
#pragma unroll
  for (int r = 0; r < 8; ++r) { mrow[r] = -3.0e38f; lrow[r] = 0.0f; }
  for (int c = 0; c < 4; ++c)
    for (int r = 0; r < 8; ++r) oacc[c][r] = 0.0f;

  const int nsteps = tBlock / 32 + 2;   // uniform over block (extra steps fully masked)
  const float scale = 0.125f;           // 1/sqrt(64)

  stage_kv(tid, ksBase, vsBase, kb, vb, 0);   // prologue -> buf 0

  for (int j = 0; j < nsteps; ++j) {
    const int kb0 = j * 32;
    const int buf = j & 1;
    wait_async0();
    __syncthreads();
    if (j + 1 < nsteps)
      stage_kv(tid, ksBase + (buf ^ 1) * 4608, vsBase + (buf ^ 1) * 5120,
               kb, vb, (j + 1) * 32);

    // S = Q * K^T : two 16x16 N-tiles, chained over K=64
    v8f sacc[2];
#pragma unroll
    for (int nt = 0; nt < 2; ++nt) {
      for (int r = 0; r < 8; ++r) sacc[nt][r] = 0.0f;
#pragma unroll
      for (int c = 0; c < 2; ++c) {
        Frag kf;
#pragma unroll
        for (int p = 0; p < 8; ++p)
          kf.u[p] = *(const unsigned int*)&Ks[buf][nt * 16 + lh][c * 32 + b_kb(p, hi)];
        sacc[nt] = __builtin_amdgcn_wmma_f32_16x16x32_bf16(
            false, qf[c].v, false, kf.v, (short)0, sacc[nt], false, false);
      }
    }

    // Online softmax per query row (row lives in one 16-lane half)
#pragma unroll
    for (int r = 0; r < 8; ++r) {
      const int m = tTile + r + hi * 8;
      float s0 = sacc[0][r] * scale;
      float s1 = sacc[1][r] * scale;
      if (kb0 + lh      > m) s0 = -3.0e38f;
      if (kb0 + 16 + lh > m) s1 = -3.0e38f;
      float mx = fmaxf(s0, s1);
      mx = fmaxf(mx, __shfl_xor(mx, 1, 32));
      mx = fmaxf(mx, __shfl_xor(mx, 2, 32));
      mx = fmaxf(mx, __shfl_xor(mx, 4, 32));
      mx = fmaxf(mx, __shfl_xor(mx, 8, 32));
      const float mnew = fmaxf(mrow[r], mx);
      const float e0 = __expf(s0 - mnew);
      const float e1 = __expf(s1 - mnew);
      float rs = e0 + e1;
      rs += __shfl_xor(rs, 1, 32);
      rs += __shfl_xor(rs, 2, 32);
      rs += __shfl_xor(rs, 4, 32);
      rs += __shfl_xor(rs, 8, 32);
      const float corr = __expf(mrow[r] - mnew);
      lrow[r] = lrow[r] * corr + rs;
      mrow[r] = mnew;
#pragma unroll
      for (int c = 0; c < 4; ++c) oacc[c][r] *= corr;
      pLds[wave][r + hi * 8][lh]      = f2bf(e0);
      pLds[wave][r + hi * 8][16 + lh] = f2bf(e1);
    }

    // Reload P as an A-fragment (16x32); per-wave buffer, no block barrier
    Frag pf;
#pragma unroll
    for (int p = 0; p < 8; ++p)
      pf.u[p] = *(const unsigned int*)&pLds[wave][lh][a_kb(p, hi)];

    // O += P * V
#pragma unroll
    for (int c = 0; c < 4; ++c) {
      Frag vf;
#pragma unroll
      for (int p = 0; p < 8; ++p)
        vf.u[p] = *(const unsigned int*)&Vs[buf][c * 16 + lh][b_kb(p, hi)];
      oacc[c] = __builtin_amdgcn_wmma_f32_16x16x32_bf16(
          false, pf.v, false, vf.v, (short)0, oacc[c], false, false);
    }
  }

  // Normalize and write y (bf16, merged-head [B*T, C])
  const int b = bh >> 4, h = bh & 15;
#pragma unroll
  for (int c = 0; c < 4; ++c) {
    const int col = h * 64 + c * 16 + lh;
#pragma unroll
    for (int r = 0; r < 8; ++r) {
      const int t = tTile + r + hi * 8;
      y[(size_t)(b * T_ + t) * C_ + col] = f2bf(oacc[c][r] / lrow[r]);
    }
  }
}

// ---------------------------------------------------------------------------
extern "C" void kernel_launch(void* const* d_in, const int* in_sizes, int n_in,
                              void* d_out, int out_size, void* d_ws, size_t ws_size,
                              hipStream_t stream) {
  const float* x  = (const float*)d_in[0];
  const float* wq = (const float*)d_in[1];
  const float* bq = (const float*)d_in[2];
  const float* wk = (const float*)d_in[3];
  const float* bk = (const float*)d_in[4];
  const float* wv = (const float*)d_in[5];
  const float* bv = (const float*)d_in[6];
  const float* wp = (const float*)d_in[7];
  const float* bp = (const float*)d_in[8];

  // Workspace (bf16): xb 16MB | wqt/wkt/wvt/wpt 2MB each | Q,K,V^T 16MB each | y 16MB
  unsigned short* xb  = (unsigned short*)d_ws;
  unsigned short* wqt = xb  + (size_t)M_ * C_;
  unsigned short* wkt = wqt + (size_t)C_ * C_;
  unsigned short* wvt = wkt + (size_t)C_ * C_;
  unsigned short* wpt = wvt + (size_t)C_ * C_;
  unsigned short* qws = wpt + (size_t)C_ * C_;
  unsigned short* kws = qws + (size_t)M_ * C_;
  unsigned short* vws = kws + (size_t)M_ * C_;
  unsigned short* yws = vws + (size_t)M_ * C_;

  convert_kernel<<<(M_ * C_) / 256, 256, 0, stream>>>(x, xb, M_ * C_);
  dim3 tg(C_ / 64, C_ / 64);
  transpose_w_kernel<<<tg, 256, 0, stream>>>(wq, wqt);
  transpose_w_kernel<<<tg, 256, 0, stream>>>(wk, wkt);
  transpose_w_kernel<<<tg, 256, 0, stream>>>(wv, wvt);
  transpose_w_kernel<<<tg, 256, 0, stream>>>(wp, wpt);

  dim3 gg(C_ / 128, M_ / 128);
  gemm_bf16_kernel<<<gg, 256, 0, stream>>>(xb, wqt, bq, qws, nullptr, 0);
  gemm_bf16_kernel<<<gg, 256, 0, stream>>>(xb, wkt, bk, kws, nullptr, 0);
  gemm_bf16_kernel<<<gg, 256, 0, stream>>>(xb, wvt, bv, vws, nullptr, 1);

  attn_kernel<<<B_ * H_ * (T_ / 64), 128, 0, stream>>>(qws, kws, vws, yws);

  gemm_bf16_kernel<<<gg, 256, 0, stream>>>(yws, wpt, bp, nullptr, (float*)d_out, 2);
}
